// CausalMultiHeadSelfAttention_80101140071013
// MI455X (gfx1250) — compile-verified, matrix operand order checked by reading, not run on hardware
//
#include <hip/hip_runtime.h>
#include <math.h>

// ---------------------------------------------------------------------------
// Types for CDNA5 WMMA (wave32, 16x16x32 f16 -> f32)
// ---------------------------------------------------------------------------
typedef _Float16 v16h __attribute__((ext_vector_type(16)));
typedef _Float16 v8h  __attribute__((ext_vector_type(8)));
typedef float    v8f  __attribute__((ext_vector_type(8)));

union H16U { v16h v; v8h h[2]; };

static __device__ __forceinline__ v8f wmma16(v16h a, v16h b, v8f c) {
    // (neg_a, A, neg_b, B, c_mod, C, reuse_a, reuse_b)
    return __builtin_amdgcn_wmma_f32_16x16x32_f16(false, a, false, b,
                                                  (short)0, c, false, false);
}

// v16h A/B fragment from two contiguous 16-byte chunks (ISA 7.12.2 layout):
// lane(lr,hi): halves[0..7] = K(base + hi*8), halves[8..15] = K(base+16+hi*8)
static __device__ __forceinline__ v16h load_frag_g(const _Float16* p0,
                                                   const _Float16* p1) {
    H16U u;
    u.h[0] = *(const v8h*)p0;
    u.h[1] = *(const v8h*)p1;
    return u.v;
}

// ---------------------------------------------------------------------------
// Problem constants
// ---------------------------------------------------------------------------
#define BATCH   4
#define SEQ     2048
#define DMODEL  1024
#define NHEADS  16
#define DHEAD   64
#define MROWS   (BATCH * SEQ)          // 8192

// ---------------------------------------------------------------------------
// fp32 -> f16 conversion (grid-stride)
// ---------------------------------------------------------------------------
__global__ void mha_cvt_f16(const float* __restrict__ src,
                            _Float16* __restrict__ dst, int n) {
    int i = blockIdx.x * blockDim.x + threadIdx.x;
    int stride = gridDim.x * blockDim.x;
    for (; i < n; i += stride) dst[i] = (_Float16)src[i];
}

// ---------------------------------------------------------------------------
// WMMA GEMM: Y[m,n] = sum_k A[m,k] * W[n,k]    (A: MxK f16, W: NxK f16)
// Block: 256 thr (8 waves). Block tile 256(M) x 64(N); wave tile 32 x 64.
// B tile (64x64 per 64-K step) staged in LDS (shared by all waves),
// double-buffered; A fragments parity-double-buffered in registers
// (no cross-buffer copies -> no v_mov / hazard-nop blocks).
// mode 0: f16 per-head [b,h,s,dh]  | mode 1: f16 per-head [b,h,dh,s]
// mode 2: f32 plain [m,n]
// Grid: (N/64, M/256) = (16, 32)
// ---------------------------------------------------------------------------
#define KB 64                           // K halves per stage (2 wmma steps)

__global__ __launch_bounds__(256)
void mha_gemm_wmma(const _Float16* __restrict__ A,
                   const _Float16* __restrict__ W,
                   void* __restrict__ out, int mode) {
    __shared__ __align__(16) _Float16 Blds[2][64][72];   // 72 = 64 + 8 pad

    const int tid  = threadIdx.x;
    const int lane = tid & 31;
    const int wv   = tid >> 5;
    const int lr   = lane & 15;
    const int hi   = lane >> 4;

    const int ntile = blockIdx.x * 64;
    const int mbase = blockIdx.y * 256 + wv * 32;

    // Cooperative B-tile load mapping: 512 x 16B chunks, 2 per thread.
    const int c0 = tid, c1 = tid + 256;
    const int r0 = c0 >> 3, cc0 = c0 & 7;
    const int r1 = c1 >> 3, cc1 = c1 & 7;
    const _Float16* wr0 = W + (size_t)(ntile + r0) * DMODEL + cc0 * 8;
    const _Float16* wr1 = W + (size_t)(ntile + r1) * DMODEL + cc1 * 8;

    const _Float16* arow0 = A + (size_t)(mbase + lr) * DMODEL;
    const _Float16* arow1 = A + (size_t)(mbase + 16 + lr) * DMODEL;

    v8f acc[2][4] = {};

    // A-fragment parity double buffer: [parity][g][s]
    v16h afr[2][2][2];

    // ---- prologue: tile 0 ----
    v8h breg0 = *(const v8h*)wr0;
    v8h breg1 = *(const v8h*)wr1;
#pragma unroll
    for (int s = 0; s < 2; ++s) {
        afr[0][0][s] = load_frag_g(arow0 + s * 32 + hi * 8,
                                   arow0 + s * 32 + 16 + hi * 8);
        afr[0][1][s] = load_frag_g(arow1 + s * 32 + hi * 8,
                                   arow1 + s * 32 + 16 + hi * 8);
    }
    *(v8h*)&Blds[0][r0][cc0 * 8] = breg0;
    *(v8h*)&Blds[0][r1][cc1 * 8] = breg1;
    __syncthreads();

    const int T = DMODEL / KB;          // 16
#pragma unroll 2
    for (int t = 0; t < T; ++t) {
        const int cur = t & 1;
        const int nxt = cur ^ 1;
        const int kkn = (t + 1) * KB;
        // ---- issue next-tile global loads (latency hidden by compute) ----
        if (t + 1 < T) {
            breg0 = *(const v8h*)(wr0 + kkn);
            breg1 = *(const v8h*)(wr1 + kkn);
#pragma unroll
            for (int s = 0; s < 2; ++s) {
                afr[nxt][0][s] =
                    load_frag_g(arow0 + kkn + s * 32 + hi * 8,
                                arow0 + kkn + s * 32 + 16 + hi * 8);
                afr[nxt][1][s] =
                    load_frag_g(arow1 + kkn + s * 32 + hi * 8,
                                arow1 + kkn + s * 32 + 16 + hi * 8);
            }
        }
        // ---- compute current tile: 16 WMMAs ----
#pragma unroll
        for (int s = 0; s < 2; ++s) {
            v16h bf[4];
#pragma unroll
            for (int nt = 0; nt < 4; ++nt)
                bf[nt] = load_frag_g(
                    &Blds[cur][nt * 16 + lr][s * 32 + hi * 8],
                    &Blds[cur][nt * 16 + lr][s * 32 + 16 + hi * 8]);
#pragma unroll
            for (int g = 0; g < 2; ++g)
#pragma unroll
                for (int nt = 0; nt < 4; ++nt)
                    acc[g][nt] = wmma16(afr[cur][g][s], bf[nt], acc[g][nt]);
        }
        // ---- stage next B tile into the other LDS buffer ----
        if (t + 1 < T) {
            *(v8h*)&Blds[nxt][r0][cc0 * 8] = breg0;
            *(v8h*)&Blds[nxt][r1][cc1 * 8] = breg1;
        }
        __syncthreads();
    }

    // ---- epilogue. C layout: VGPR j -> row (hi*8+j), col = lr ----
#pragma unroll
    for (int g = 0; g < 2; ++g) {
#pragma unroll
        for (int nt = 0; nt < 4; ++nt) {
#pragma unroll
            for (int j = 0; j < 8; ++j) {
                const int m = mbase + g * 16 + hi * 8 + j;
                const int n = ntile + nt * 16 + lr;
                const float val = acc[g][nt][j];
                if (mode == 2) {
                    ((float*)out)[(size_t)m * DMODEL + n] = val;
                } else {
                    const int b  = m >> 11;
                    const int s  = m & (SEQ - 1);
                    const int h  = n >> 6;
                    const int dh = n & (DHEAD - 1);
                    if (mode == 0) {
                        ((_Float16*)out)[((((size_t)b * NHEADS + h) * SEQ + s)
                                          << 6) + dh] = (_Float16)val;
                    } else {
                        ((_Float16*)out)[(((size_t)b * NHEADS + h) * DHEAD
                                          + dh) * SEQ + s] = (_Float16)val;
                    }
                }
            }
        }
    }
}

// ---------------------------------------------------------------------------
// Fused causal flash-attention per (b,h).
// Qh, Kh: [B,H,S,DH] f16 ; Vt: [B,H,DH,S] f16 ; Ctx out: [B,S,H*DH] f16.
// One wave owns 16 query rows; iterates 32-key tiles up to the causal limit.
// K fragments for tile t+1 are prefetched into a parity buffer during tile
// t's compute; V fragments are issued before the score WMMAs so the softmax
// VALU (co-executing with XDL WMMA) hides them.
// Grid: (S/16/8, B*H), block 256.
// ---------------------------------------------------------------------------
struct KFrag { v16h f00, f01, f10, f11; };

static __device__ __forceinline__ KFrag load_kfrags(const _Float16* Kp,
                                                    int kb, int lr, int hi) {
    const _Float16* kr0 = Kp + (size_t)(kb + lr) * DHEAD;
    const _Float16* kr1 = Kp + (size_t)(kb + 16 + lr) * DHEAD;
    KFrag kf;
    kf.f00 = load_frag_g(kr0 + hi * 8,      kr0 + 16 + hi * 8);
    kf.f01 = load_frag_g(kr0 + 32 + hi * 8, kr0 + 48 + hi * 8);
    kf.f10 = load_frag_g(kr1 + hi * 8,      kr1 + 16 + hi * 8);
    kf.f11 = load_frag_g(kr1 + 32 + hi * 8, kr1 + 48 + hi * 8);
    return kf;
}

__global__ __launch_bounds__(256)
void mha_attn_wmma(const _Float16* __restrict__ Qh,
                   const _Float16* __restrict__ Kh,
                   const _Float16* __restrict__ Vt,
                   _Float16* __restrict__ Ctx) {
    __shared__ __align__(16) _Float16 ldsP[8][16][32];   // 1KB per wave

    const int lane = threadIdx.x & 31;
    const int wv   = threadIdx.x >> 5;
    const int lr   = lane & 15;
    const int hi   = lane >> 4;

    const int bh    = blockIdx.y;                 // b*NHEADS + h
    const int qbase = (blockIdx.x * 8 + wv) * 16;

    const _Float16* Qp = Qh + (size_t)bh * SEQ * DHEAD;
    const _Float16* Kp = Kh + (size_t)bh * SEQ * DHEAD;
    const _Float16* Vp = Vt + (size_t)bh * DHEAD * SEQ;

    const _Float16* qrow = Qp + (size_t)(qbase + lr) * DHEAD;
    const v16h qf0 = load_frag_g(qrow + hi * 8,      qrow + 16 + hi * 8);
    const v16h qf1 = load_frag_g(qrow + 32 + hi * 8, qrow + 48 + hi * 8);

    const float NEGINF = -__builtin_inff();
    const float sc = 0.18033688f;   // (1/sqrt(64)) * log2(e): softmax in log2

    float mrun[8], lrun[8];
#pragma unroll
    for (int j = 0; j < 8; ++j) { mrun[j] = NEGINF; lrun[j] = 0.0f; }
    v8f acc[4] = {};

    const int nkt = (qbase + 16 + 31) >> 5;       // # of 32-key tiles (causal)

    KFrag kfr[2];
    kfr[0] = load_kfrags(Kp, 0, lr, hi);

#pragma unroll 2
    for (int t = 0; t < nkt; ++t) {
        const int kb  = t * 32;
        const int cur = t & 1;

        // ---- prefetch next K tile + this tile's V fragments ----
        if (t + 1 < nkt) kfr[cur ^ 1] = load_kfrags(Kp, kb + 32, lr, hi);
        v16h vf[4];
#pragma unroll
        for (int nt = 0; nt < 4; ++nt) {
            const _Float16* vrow = Vp + (size_t)(nt * 16 + lr) * SEQ;
            vf[nt] = load_frag_g(vrow + kb + hi * 8, vrow + kb + 16 + hi * 8);
        }

        // ---- scores S = Q . K^T (32 keys = two 16-key column tiles) ----
        v8f s0 = {}, s1 = {};
        s0 = wmma16(qf0, kfr[cur].f00, s0);
        s0 = wmma16(qf1, kfr[cur].f01, s0);
        s1 = wmma16(qf0, kfr[cur].f10, s1);
        s1 = wmma16(qf1, kfr[cur].f11, s1);

        // ---- scale + causal mask + online softmax ----
        float sa[8], sb[8], tm[8];
        const int key0 = kb + lr;
        const int key1 = kb + 16 + lr;
#pragma unroll
        for (int j = 0; j < 8; ++j) {
            const int row = qbase + hi * 8 + j;
            float a = s0[j] * sc; if (key0 > row) a = NEGINF;
            float b = s1[j] * sc; if (key1 > row) b = NEGINF;
            sa[j] = a; sb[j] = b;
            tm[j] = fmaxf(a, b);
        }
#pragma unroll
        for (int j = 0; j < 8; ++j) {           // row-max across 16-lane half
            float v = tm[j];
            v = fmaxf(v, __shfl_xor(v, 1, 32));
            v = fmaxf(v, __shfl_xor(v, 2, 32));
            v = fmaxf(v, __shfl_xor(v, 4, 32));
            v = fmaxf(v, __shfl_xor(v, 8, 32));
            tm[j] = v;
        }

        float p0[8], p1[8], alpha[8];
#pragma unroll
        for (int j = 0; j < 8; ++j) {
            const float mnew = fmaxf(mrun[j], tm[j]);
            alpha[j] = exp2f(mrun[j] - mnew);   // exp2(-inf - x) = 0
            p0[j] = exp2f(sa[j] - mnew);
            p1[j] = exp2f(sb[j] - mnew);
            mrun[j] = mnew;
        }
#pragma unroll
        for (int j = 0; j < 8; ++j) {           // row-sum across half
            float v = p0[j] + p1[j];
            v += __shfl_xor(v, 1, 32);
            v += __shfl_xor(v, 2, 32);
            v += __shfl_xor(v, 4, 32);
            v += __shfl_xor(v, 8, 32);
            lrun[j] = lrun[j] * alpha[j] + v;
        }
#pragma unroll
        for (int nt = 0; nt < 4; ++nt)
#pragma unroll
            for (int j = 0; j < 8; ++j) acc[nt][j] *= alpha[j];

        // ---- C-layout P -> A-layout via per-wave LDS bounce ----
#pragma unroll
        for (int j = 0; j < 8; ++j) {
            ldsP[wv][hi * 8 + j][lr]      = (_Float16)p0[j];
            ldsP[wv][hi * 8 + j][16 + lr] = (_Float16)p1[j];
        }
        asm volatile("s_wait_dscnt 0" ::: "memory");   // per-wave LDS RAW

        H16U pu;
        pu.h[0] = *(const v8h*)&ldsP[wv][lr][hi * 8];
        pu.h[1] = *(const v8h*)&ldsP[wv][lr][16 + hi * 8];
        const v16h pf = pu.v;

        // ---- acc += P(16x32) . V(32x64)  (V stored [dh][s]) ----
#pragma unroll
        for (int nt = 0; nt < 4; ++nt)
            acc[nt] = wmma16(pf, vf[nt], acc[nt]);
    }

    // ---- finalize: divide by l, store context [b, s, h*64+dh] f16 ----
    const int b = bh >> 4;
    const int h = bh & 15;
#pragma unroll
    for (int nt = 0; nt < 4; ++nt) {
#pragma unroll
        for (int j = 0; j < 8; ++j) {
            const int s  = qbase + hi * 8 + j;
            const int dh = nt * 16 + lr;
            const float o = acc[nt][j] / lrun[j];
            Ctx[((size_t)(b * SEQ + s)) * DMODEL + h * DHEAD + dh] =
                (_Float16)o;
        }
    }
}

// ---------------------------------------------------------------------------
// Host-side launch
// ---------------------------------------------------------------------------
extern "C" void kernel_launch(void* const* d_in, const int* in_sizes, int n_in,
                              void* d_out, int out_size, void* d_ws,
                              size_t ws_size, hipStream_t stream) {
    const float* X  = (const float*)d_in[0];
    const float* WQ = (const float*)d_in[1];
    const float* WK = (const float*)d_in[2];
    const float* WV = (const float*)d_in[3];
    const float* WO = (const float*)d_in[4];

    char* ws = (char*)d_ws;
    size_t off = 0;
    _Float16* Xh  = (_Float16*)(ws + off); off += (size_t)MROWS * DMODEL * 2;
    _Float16* WQh = (_Float16*)(ws + off); off += (size_t)DMODEL * DMODEL * 2;
    _Float16* WKh = (_Float16*)(ws + off); off += (size_t)DMODEL * DMODEL * 2;
    _Float16* WVh = (_Float16*)(ws + off); off += (size_t)DMODEL * DMODEL * 2;
    _Float16* WOh = (_Float16*)(ws + off); off += (size_t)DMODEL * DMODEL * 2;
    _Float16* Qh  = (_Float16*)(ws + off); off += (size_t)MROWS * DMODEL * 2;
    _Float16* Kh  = (_Float16*)(ws + off); off += (size_t)MROWS * DMODEL * 2;
    _Float16* Vt  = (_Float16*)(ws + off); off += (size_t)MROWS * DMODEL * 2;
    _Float16* Ctx = (_Float16*)(ws + off); off += (size_t)MROWS * DMODEL * 2;

    const int nX = MROWS * DMODEL;
    const int nW = DMODEL * DMODEL;
    mha_cvt_f16<<<2048, 256, 0, stream>>>(X,  Xh,  nX);
    mha_cvt_f16<<<1024, 256, 0, stream>>>(WQ, WQh, nW);
    mha_cvt_f16<<<1024, 256, 0, stream>>>(WK, WKh, nW);
    mha_cvt_f16<<<1024, 256, 0, stream>>>(WV, WVh, nW);
    mha_cvt_f16<<<1024, 256, 0, stream>>>(WO, WOh, nW);

    // Projections: grid (N/64, M/256) = (16, 32)
    dim3 ggrid(DMODEL / 64, MROWS / 256);
    mha_gemm_wmma<<<ggrid, 256, 0, stream>>>(Xh, WQh, (void*)Qh, 0);
    mha_gemm_wmma<<<ggrid, 256, 0, stream>>>(Xh, WKh, (void*)Kh, 0);
    mha_gemm_wmma<<<ggrid, 256, 0, stream>>>(Xh, WVh, (void*)Vt, 1);

    // Attention: grid (S/16/8, B*H) = (16, 64)
    dim3 agrid(SEQ / 16 / 8, BATCH * NHEADS);
    mha_attn_wmma<<<agrid, 256, 0, stream>>>(Qh, Kh, Vt, Ctx);

    // Output projection -> fp32 d_out
    mha_gemm_wmma<<<ggrid, 256, 0, stream>>>(Ctx, WOh, d_out, 2);
}
